// MultiLatentHeadAttention_70188355551255
// MI455X (gfx1250) — matrix operand order repaired and checked by reading, not compiled, not run on hardware
//
#include <hip/hip_runtime.h>
#include <hip/hip_bf16.h>

// ---------------------------------------------------------------------------
// MLA (multi-latent attention) for gfx1250: bf16 WMMA pipeline, wave32,
// pre-transposed bf16 weights, double-buffered TDM staging of K/V tiles.
// Shapes: B=2, T=2048, C=2048, H=16, hd=128, L=512.
// ---------------------------------------------------------------------------

typedef __attribute__((ext_vector_type(16))) __bf16 v16bf;
typedef __attribute__((ext_vector_type(8)))  __bf16 v8bf;
typedef __attribute__((ext_vector_type(8)))  float  v8f;
typedef __attribute__((ext_vector_type(4)))  float  f32x4;
typedef __attribute__((ext_vector_type(4)))  unsigned int u32x4;
typedef __attribute__((ext_vector_type(8)))  int    i32x8;
typedef __attribute__((ext_vector_type(4)))  int    i32x4;

#define NB   2
#define NT   2048
#define NC   2048
#define NH   16
#define NHD  128
#define NL   512
#define NBT  (NB * NT)

#define KTILE    32           // keys per flash iteration
#define LDS_LDK  136          // K tile: 128 bf16 row + 16B pad (272B rows)
#define LDS_LDV  40           // V^T tile: 32 bf16 row + 16B pad (80B rows)

__device__ inline v8f wmma_bf16(v16bf a, v16bf b, v8f c) {
  return __builtin_amdgcn_wmma_f32_16x16x32_bf16(
      false, a, false, b, (short)0, c, false, false);
}

// ---- A fragment 16x32 from f32 row-major: 4x b128 loads per lane.
__device__ inline v16bf load_a_f32(const float* __restrict__ A, int lda, int m0, int k0) {
  int lane = threadIdx.x & 31;
  int m = lane & 15, kh = lane >> 4;
  const float* row = A + (size_t)(m0 + m) * lda + k0 + kh * 8;
  f32x4 c0 = *(const f32x4*)(row);
  f32x4 c1 = *(const f32x4*)(row + 4);
  f32x4 c2 = *(const f32x4*)(row + 16);
  f32x4 c3 = *(const f32x4*)(row + 20);
  v16bf a;
#pragma unroll
  for (int j = 0; j < 4; ++j) {
    a[j]      = (__bf16)c0[j];
    a[4 + j]  = (__bf16)c1[j];
    a[8 + j]  = (__bf16)c2[j];
    a[12 + j] = (__bf16)c3[j];
  }
  return a;
}

// ---- A fragment 16x32 from bf16 row-major: 2x b128 loads per lane.
__device__ inline v16bf load_a_bf16(const __bf16* __restrict__ A, int lda, int m0, int k0) {
  int lane = threadIdx.x & 31;
  int m = lane & 15, kh = lane >> 4;
  const __bf16* row = A + (size_t)(m0 + m) * lda + k0 + kh * 8;
  v8bf c0 = *(const v8bf*)(row);
  v8bf c1 = *(const v8bf*)(row + 16);
  v16bf a;
#pragma unroll
  for (int j = 0; j < 8; ++j) { a[j] = c0[j]; a[8 + j] = c1[j]; }
  return a;
}

// ---- B fragment from N-major (transposed) bf16: B(k,n) = M[n0+n][k0+k].
// Table k order (k = 8h+s, 16+8h+s-8): 2x b128 loads per lane.
__device__ inline v16bf load_bt(const __bf16* __restrict__ M, int ldm, int n0, int k0) {
  int lane = threadIdx.x & 31;
  int n = lane & 15, kh = lane >> 4;
  const __bf16* row = M + (size_t)(n0 + n) * ldm + k0 + kh * 8;
  v8bf c0 = *(const v8bf*)(row);
  v8bf c1 = *(const v8bf*)(row + 16);
  v16bf b;
#pragma unroll
  for (int j = 0; j < 8; ++j) { b[j] = c0[j]; b[8 + j] = c1[j]; }
  return b;
}

// RoPE: inv_freq = THETA^(-2i/hd) = exp2(-(2i/hd)*log2(THETA)), THETA=1e5.
__device__ inline float rope_angle(int t, int i) {
  const float k = -2.0f * 16.609640474436812f / (float)NHD;
  return (float)t * exp2f((float)i * k);
}

// ---- Tensor Data Mover: 2D bf16 tile (tile_d0 x tile_d1) from global
// row-major (stride0 elements) into LDS, padded pad_amt dwords per interval.
__device__ inline void tdm_load_tile_bf16(unsigned lds_off, const __bf16* gptr,
                                          unsigned tile_d0, unsigned tile_d1,
                                          unsigned stride0,
                                          unsigned pad_interval_code,
                                          unsigned pad_amount_code) {
  unsigned long long ga = (unsigned long long)(size_t)gptr;
  u32x4 g0;
  g0[0] = 1u;                                   // count=1 (valid user D#)
  g0[1] = lds_off;                              // lds_addr (bytes)
  g0[2] = (unsigned)(ga & 0xFFFFFFFFu);         // global_addr[31:0]
  g0[3] = (unsigned)((ga >> 32) & 0x01FFFFFFu)  // global_addr[56:32]
          | (2u << 30);                         // type = 2 (image)
  i32x8 g1;
  g1[0] = (int)((1u << 16) | (1u << 20)                   // data_size=2B, pad_en
          | (pad_interval_code << 22) | (pad_amount_code << 25));
  g1[1] = (int)((tile_d0 & 0xFFFFu) << 16);               // tensor_dim0[15:0]
  g1[2] = (int)(((tile_d0 >> 16) & 0xFFFFu)
          | ((tile_d1 & 0xFFFFu) << 16));                 // tensor_dim1[15:0]
  g1[3] = (int)(((tile_d1 >> 16) & 0xFFFFu)
          | ((tile_d0 & 0xFFFFu) << 16));                 // tile_dim0
  g1[4] = (int)(tile_d1 & 0xFFFFu);                       // tile_dim1
  g1[5] = (int)stride0;                                   // tensor_dim0_stride
  g1[6] = 0;
  g1[7] = 0;
  i32x4 z4 = {0, 0, 0, 0};
  i32x8 z8 = {0, 0, 0, 0, 0, 0, 0, 0};
  __builtin_amdgcn_tensor_load_to_lds(g0, g1, z4, z4, z8, 0);
}

// ===========================================================================
// Kernel 0: transpose + f32->bf16 weights: src (K x N) -> dst (N x K), batched.
// ===========================================================================
__global__ void k_transpose_bf16(const float* __restrict__ src,
                                 __bf16* __restrict__ dst, int K, int N) {
  __shared__ float tile[32][33];
  src += (size_t)blockIdx.z * K * N;
  dst += (size_t)blockIdx.z * K * N;
  int k0 = blockIdx.x * 32, n0 = blockIdx.y * 32;
  int tx = threadIdx.x & 31, ty = threadIdx.x >> 5;   // 32 x 8
#pragma unroll
  for (int r = 0; r < 32; r += 8)
    tile[ty + r][tx] = src[(size_t)(k0 + ty + r) * N + n0 + tx];
  __syncthreads();
#pragma unroll
  for (int r = 0; r < 32; r += 8)
    dst[(size_t)(n0 + ty + r) * K + k0 + tx] = (__bf16)tile[tx][ty + r];
}

// ===========================================================================
// Kernel 1: q = x @ Wq with fused RoPE -> bf16 (B,H,T,hd).
// ===========================================================================
__global__ void k_proj_q_rope(const float* __restrict__ x,
                              const __bf16* __restrict__ WqT,   // (H*hd) x C
                              __bf16* __restrict__ qr) {
  __shared__ float sq[16][NHD];
  int w = threadIdx.x >> 5;
  int r0 = blockIdx.x * 16;
  int h  = blockIdx.y;
  v8f c = {};
  for (int k0 = 0; k0 < NC; k0 += 32)
    c = wmma_bf16(load_a_f32(x, NC, r0, k0),
                  load_bt(WqT, NC, h * NHD + w * 16, k0), c);
  int lane = threadIdx.x & 31, n = lane & 15, mh = lane >> 4;
#pragma unroll
  for (int v = 0; v < 8; ++v) sq[v + 8 * mh][w * 16 + n] = c[v];
  __syncthreads();
  for (int e = threadIdx.x; e < 16 * NHD; e += 256) {
    int row = e >> 7, d = e & (NHD - 1);
    int gr = r0 + row;
    int bb = gr >> 11, t = gr & (NT - 1);
    int i  = d & 63;
    float fr = rope_angle(t, i);
    float cs = __cosf(fr), sn = __sinf(fr);
    float x1 = sq[row][i], x2 = sq[row][i + 64];
    float o = (d < 64) ? (x1 * cs - x2 * sn) : (x1 * sn + x2 * cs);
    qr[(((size_t)bb * NH + h) * NT + t) * NHD + d] = (__bf16)o;
  }
}

// ===========================================================================
// Kernel 2: kv = (x @ Wkv) * 1/sqrt(H), fp32 (BT x L).
// ===========================================================================
__global__ void k_proj_kv(const float* __restrict__ x,
                          const __bf16* __restrict__ WkvT,      // L x C
                          float* __restrict__ kv) {
  int w = threadIdx.x >> 5;
  int r0 = blockIdx.x * 16;
  int n0 = blockIdx.y * 128 + w * 16;
  v8f c = {};
  for (int k0 = 0; k0 < NC; k0 += 32)
    c = wmma_bf16(load_a_f32(x, NC, r0, k0), load_bt(WkvT, NC, n0, k0), c);
  int lane = threadIdx.x & 31, n = lane & 15, mh = lane >> 4;
#pragma unroll
  for (int v = 0; v < 8; ++v)
    kv[(size_t)(r0 + v + 8 * mh) * NL + n0 + n] = c[v] * 0.25f;
}

// ===========================================================================
// Kernel 3: per-head k = kv @ Wk_up[h] (+RoPE) -> (B,H,T,hd);
//           v = kv @ Wv_up[h] -> TRANSPOSED (B,H,hd,T).
// ===========================================================================
__global__ void k_up_kv(const float* __restrict__ kv,
                        const __bf16* __restrict__ WkupT,   // [h] hd x L
                        const __bf16* __restrict__ WvupT,   // [h] hd x L
                        __bf16* __restrict__ kb,
                        __bf16* __restrict__ vbT) {
  __shared__ float sk[16][NHD];
  __shared__ float sv[16][NHD + 1];
  int w = threadIdx.x >> 5;
  int r0 = blockIdx.x * 16;
  int h  = blockIdx.y;
  const __bf16* Wk = WkupT + (size_t)h * NHD * NL;
  const __bf16* Wv = WvupT + (size_t)h * NHD * NL;
  v8f ck = {}, cv = {};
  for (int k0 = 0; k0 < NL; k0 += 32) {
    v16bf a = load_a_f32(kv, NL, r0, k0);
    ck = wmma_bf16(a, load_bt(Wk, NL, w * 16, k0), ck);
    cv = wmma_bf16(a, load_bt(Wv, NL, w * 16, k0), cv);
  }
  int lane = threadIdx.x & 31, n = lane & 15, mh = lane >> 4;
#pragma unroll
  for (int v = 0; v < 8; ++v) {
    sk[v + 8 * mh][w * 16 + n] = ck[v];
    sv[v + 8 * mh][w * 16 + n] = cv[v];
  }
  __syncthreads();
  for (int e = threadIdx.x; e < 16 * NHD; e += 256) {
    int row = e >> 7, d = e & (NHD - 1);
    int gr = r0 + row;
    int bb = gr >> 11, t = gr & (NT - 1);
    int i  = d & 63;
    float fr = rope_angle(t, i);
    float cs = __cosf(fr), sn = __sinf(fr);
    float x1 = sk[row][i], x2 = sk[row][i + 64];
    float o = (d < 64) ? (x1 * cs - x2 * sn) : (x1 * sn + x2 * cs);
    kb[(((size_t)bb * NH + h) * NT + t) * NHD + d] = (__bf16)o;
  }
  // V transposed store: (B,H,hd,T); tl fastest -> coalesced along T.
  for (int e = threadIdx.x; e < 16 * NHD; e += 256) {
    int d = e >> 4, tl = e & 15;
    int gr = r0 + tl;
    int bb = gr >> 11, t = gr & (NT - 1);
    vbT[(((size_t)bb * NH + h) * NHD + d) * NT + t] = (__bf16)sv[tl][d];
  }
}

// ===========================================================================
// Kernel 4: flash attention, double-buffered TDM staging of K and V^T tiles.
// grid (T/128, H, B), block 256 (8 waves). Wave w: 16 q-rows x hd=128.
// ===========================================================================
__global__ void k_flash(const __bf16* __restrict__ qr,
                        const __bf16* __restrict__ kbuf,
                        const __bf16* __restrict__ vbT,
                        __bf16* __restrict__ attn) {
  __shared__ __bf16 Kt[2][KTILE][LDS_LDK];   // keys x d (padded)
  __shared__ __bf16 Vt[2][NHD][LDS_LDV];     // d x keys (padded)
  __shared__ __bf16 Pbuf[8][16][32];

  int w = threadIdx.x >> 5, lane = threadIdx.x & 31;
  int n = lane & 15, mh = lane >> 4;
  int h = blockIdx.y, b = blockIdx.z;
  int r0 = blockIdx.x * 128 + w * 16;

  const __bf16* Q = qr   + ((size_t)b * NH + h) * NT * NHD;
  const __bf16* K = kbuf + ((size_t)b * NH + h) * NT * NHD;
  const __bf16* V = vbT  + ((size_t)b * NH + h) * NHD * NT;  // d-major

  v16bf qf[4];
#pragma unroll
  for (int dc = 0; dc < 4; ++dc) qf[dc] = load_a_bf16(Q, NHD, r0, dc * 32);

  v8f zero = {};
  v8f o[8];
  float m_r[8], l_r[8];
#pragma unroll
  for (int i = 0; i < 8; ++i) { o[i] = zero; m_r[i] = -1e30f; l_r[i] = 0.0f; }

  const float sscale = 0.08838834764831845f;  // 1/sqrt(128)
  unsigned kt_off[2] = {(unsigned)(size_t)&Kt[0][0][0], (unsigned)(size_t)&Kt[1][0][0]};
  unsigned vt_off[2] = {(unsigned)(size_t)&Vt[0][0][0], (unsigned)(size_t)&Vt[1][0][0]};
  int kmax = blockIdx.x * 128 + 128;          // block-uniform causal bound

  // Prefetch tile 0 into buffer 0 (K: 32x128 rows; V^T: 128x32 rows).
  if (w == 0) {
    tdm_load_tile_bf16(kt_off[0], K, NHD, KTILE, NHD, 5, 3);
    tdm_load_tile_bf16(vt_off[0], V, KTILE, NHD, NT, 3, 3);
  }

  int ib = 0;
  for (int kb = 0; kb < kmax; kb += KTILE, ib ^= 1) {
    if (w == 0) {
      if (kb + KTILE < kmax) {
        // issue next pair, then wait until only those 2 remain outstanding
        tdm_load_tile_bf16(kt_off[ib ^ 1], K + (size_t)(kb + KTILE) * NHD,
                           NHD, KTILE, NHD, 5, 3);
        tdm_load_tile_bf16(vt_off[ib ^ 1], V + (size_t)(kb + KTILE),
                           KTILE, NHD, NT, 3, 3);
        __builtin_amdgcn_s_wait_tensorcnt(2);
      } else {
        __builtin_amdgcn_s_wait_tensorcnt(0);
      }
    }
    __syncthreads();

    // --- S = Q * K^T (two 16-key subtiles)
    v8f s0 = zero, s1 = zero;
#pragma unroll
    for (int dc = 0; dc < 4; ++dc) {
      s0 = wmma_bf16(qf[dc], load_bt(&Kt[ib][0][0], LDS_LDK, 0,  dc * 32), s0);
      s1 = wmma_bf16(qf[dc], load_bt(&Kt[ib][0][0], LDS_LDK, 16, dc * 32), s1);
    }
    // --- scale + causal mask + online softmax
#pragma unroll
    for (int v = 0; v < 8; ++v) {
      int q_idx = r0 + v + 8 * mh;
      float a0 = (kb + n      <= q_idx) ? s0[v] * sscale : -1e30f;
      float a1 = (kb + 16 + n <= q_idx) ? s1[v] * sscale : -1e30f;
      float mx = fmaxf(a0, a1);
#pragma unroll
      for (int off = 8; off >= 1; off >>= 1) mx = fmaxf(mx, __shfl_xor(mx, off, 32));
      float mn = fmaxf(m_r[v], mx);
      float p0 = __expf(a0 - mn);
      float p1 = __expf(a1 - mn);
      Pbuf[w][v + 8 * mh][n]      = (__bf16)p0;
      Pbuf[w][v + 8 * mh][16 + n] = (__bf16)p1;
      float rs = p0 + p1;
#pragma unroll
      for (int off = 8; off >= 1; off >>= 1) rs += __shfl_xor(rs, off, 32);
      float sc = __expf(m_r[v] - mn);
      l_r[v] = l_r[v] * sc + rs;
      m_r[v] = mn;
#pragma unroll
      for (int ds = 0; ds < 8; ++ds) o[ds][v] *= sc;
    }
    // --- O += P @ V (P via LDS round-trip; V^T fragments contiguous)
    v16bf pf = load_a_bf16(&Pbuf[w][0][0], 32, 0, 0);
#pragma unroll
    for (int ds = 0; ds < 8; ++ds)
      o[ds] = wmma_bf16(pf, load_bt(&Vt[ib][0][0], LDS_LDV, ds * 16, 0), o[ds]);

    __syncthreads();  // tiles fully consumed before next DMA overwrites them
  }

  float inv_l[8];
#pragma unroll
  for (int v = 0; v < 8; ++v) inv_l[v] = 1.0f / l_r[v];
#pragma unroll
  for (int ds = 0; ds < 8; ++ds) {
#pragma unroll
    for (int v = 0; v < 8; ++v) {
      int t = r0 + v + 8 * mh;
      attn[((size_t)b * NT + t) * NC + h * NHD + ds * 16 + n] =
          (__bf16)(o[ds][v] * inv_l[v]);
    }
  }
}

// ===========================================================================
// Kernel 5: out = attn_out @ Wo, fp32 output.
// ===========================================================================
__global__ void k_out_proj(const __bf16* __restrict__ attn,
                           const __bf16* __restrict__ WoT,   // C x C (N-major)
                           float* __restrict__ out) {
  int w = threadIdx.x >> 5;
  int r0 = blockIdx.x * 16;
  int n0 = blockIdx.y * 128 + w * 16;
  v8f c = {};
  for (int k0 = 0; k0 < NC; k0 += 32)
    c = wmma_bf16(load_a_bf16(attn, NC, r0, k0), load_bt(WoT, NC, n0, k0), c);
  int lane = threadIdx.x & 31, n = lane & 15, mh = lane >> 4;
#pragma unroll
  for (int v = 0; v < 8; ++v)
    out[(size_t)(r0 + v + 8 * mh) * NC + n0 + n] = c[v];
}

// ===========================================================================
extern "C" void kernel_launch(void* const* d_in, const int* in_sizes, int n_in,
                              void* d_out, int out_size, void* d_ws, size_t ws_size,
                              hipStream_t stream) {
  (void)in_sizes; (void)n_in; (void)out_size; (void)ws_size;
  const float* x     = (const float*)d_in[0];
  const float* Wq    = (const float*)d_in[1];
  const float* Wkv   = (const float*)d_in[2];
  const float* Wk_up = (const float*)d_in[3];
  const float* Wv_up = (const float*)d_in[4];
  const float* Wo    = (const float*)d_in[5];
  float* out = (float*)d_out;

  char* ws = (char*)d_ws;
  size_t off = 0;
  auto carve = [&](size_t bytes) {
    void* p = ws + off;
    off = (off + bytes + 255) & ~(size_t)255;
    return p;
  };
  const size_t nBHTD = (size_t)NB * NH * NT * NHD;
  __bf16* qr    = (__bf16*)carve(nBHTD * 2);
  __bf16* kbuf  = (__bf16*)carve(nBHTD * 2);
  __bf16* vbT   = (__bf16*)carve(nBHTD * 2);
  float*  kv    = (float*) carve((size_t)NBT * NL * 4);
  __bf16* attn  = (__bf16*)carve((size_t)NBT * NC * 2);
  __bf16* WqT   = (__bf16*)carve((size_t)NC * NC * 2);
  __bf16* WkvT  = (__bf16*)carve((size_t)NC * NL * 2);
  __bf16* WkupT = (__bf16*)carve((size_t)NH * NL * NHD * 2);
  __bf16* WvupT = (__bf16*)carve((size_t)NH * NL * NHD * 2);
  __bf16* WoT   = (__bf16*)carve((size_t)NC * NC * 2);

  dim3 blk(256);
  // Weight prep (transpose + bf16 convert)
  k_transpose_bf16<<<dim3(64, 64, 1),  blk, 0, stream>>>(Wq,    WqT,   NC, NC);
  k_transpose_bf16<<<dim3(64, 16, 1),  blk, 0, stream>>>(Wkv,   WkvT,  NC, NL);
  k_transpose_bf16<<<dim3(16, 4, NH),  blk, 0, stream>>>(Wk_up, WkupT, NL, NHD);
  k_transpose_bf16<<<dim3(16, 4, NH),  blk, 0, stream>>>(Wv_up, WvupT, NL, NHD);
  k_transpose_bf16<<<dim3(64, 64, 1),  blk, 0, stream>>>(Wo,    WoT,   NC, NC);
  // Pipeline
  k_proj_q_rope<<<dim3(NBT / 16, NH), blk, 0, stream>>>(x, WqT, qr);
  k_proj_kv    <<<dim3(NBT / 16, NL / 128), blk, 0, stream>>>(x, WkvT, kv);
  k_up_kv      <<<dim3(NBT / 16, NH), blk, 0, stream>>>(kv, WkupT, WvupT, kbuf, vbT);
  k_flash      <<<dim3(NT / 128, NH, NB), blk, 0, stream>>>(qr, kbuf, vbT, attn);
  k_out_proj   <<<dim3(NBT / 16, NC / 128), blk, 0, stream>>>(attn, WoT, out);
}
